// CombinedBasisLayer_60885456388460
// MI455X (gfx1250) — compile-verified, matrix operand order checked by reading, not run on hardware
//
#include <hip/hip_runtime.h>
#include <hip/hip_bf16.h>
#include <math.h>

typedef __attribute__((ext_vector_type(16))) _Float16 v16h;
typedef __attribute__((ext_vector_type(4)))  _Float16 v4h;
typedef __attribute__((ext_vector_type(8)))  float    v8f;

#define INPUT_DIM 32
#define OUTPUT_DIM 64
#define GRIDSIZE 300
#define RBF_GRIDS 8
#define TOTAL_CH 320        // 64 + 8*32
#define BB 4
#define HH 32
#define WW 32
#define NPTS (BB*HH*WW)     // 4096
#define BN_EPS 1e-5f

#define NCHUNK 10           // ceil(300/32)
// packed B: [i][chunk][t][jtile][lane][e] halves
#define PACK_ELEMS ((size_t)2 * INPUT_DIM * NCHUNK * 4 * 32 * 16)   // 1,310,720

// ---------------------------------------------------------------------------
// Kernel 0: one-shot pack of fouriercoeffs (f32) into f16 WMMA B-fragment
// layout. Zero-pads g >= 300 so the main kernel needs no tail masking.
//   packed[(((i*10+chunk)*2 + t)*4 + jtile)*512 + lane*16 + e]
//     = fc[t][jtile*16 + (lane&15)][i][chunk*32 + K(e,lane>>4)]
// ---------------------------------------------------------------------------
__global__ __launch_bounds__(256) void pack_b_kernel(
    const float* __restrict__ fc, _Float16* __restrict__ packed)
{
    const int tid = blockIdx.x * 256 + threadIdx.x;
    if (tid >= (int)PACK_ELEMS) return;
    const int e     = tid & 15;
    const int lane  = (tid >> 4) & 31;
    const int jtile = (tid >> 9) & 3;
    const int t     = (tid >> 11) & 1;
    const int rest  = tid >> 12;          // i*10 + chunk
    const int chunk = rest % NCHUNK;
    const int i     = rest / NCHUNK;
    const int half  = lane >> 4;
    const int j     = jtile * 16 + (lane & 15);
    const int K     = e + half * 8 + ((e >= 8) ? 8 : 0);
    const int g     = chunk * 32 + K;
    const float v = (g < GRIDSIZE)
        ? fc[(((size_t)t * OUTPUT_DIM + j) * INPUT_DIM + i) * GRIDSIZE + g]
        : 0.0f;
    packed[tid] = (_Float16)v;
}

// ---------------------------------------------------------------------------
// Kernel 1: Fourier-KAN einsum on the matrix pipe.
// Block = 128 threads (4 waves) owning one 16-row N tile; wave w owns J tile w.
// Per (i, chunk): 128 threads cooperatively generate the shared 16x32 trig
// A-fragments (cos+sin) into double-buffered LDS (phase recurrence in
// revolutions -> raw v_cos_f32/v_sin_f32, no per-element int->float or 1/2pi
// muls), then each wave does 2x2 ds_load_b128 (A), 2x2 global_load_b128
// (pre-packed f16 B) and 2x v_wmma_f32_16x16x32_f16.
// ---------------------------------------------------------------------------
__global__ __launch_bounds__(128) void fourier_wmma_kernel(
    const float* __restrict__ x, const _Float16* __restrict__ bp,
    const float* __restrict__ fwp, float* __restrict__ combined)
{
    __shared__ __align__(32) _Float16 lds_a[2][2][512];  // [buf][cos/sin][lane*16+e]

    const int tid  = threadIdx.x;
    const int lane = tid & 31;
    const int wave = tid >> 5;            // 0..3 -> J tile
    const int ntile = blockIdx.x;         // 0..255 -> N tile
    const int mrow = lane & 15;
    const int half = lane >> 4;

    // ---- generator role: thread covers row (tid&15), K quad (tid>>4)&7 ----
    const int grow = tid & 15;
    const int kg   = (tid >> 4) & 7;
    const int K0   = kg * 4;                                  // K0..K0+3
    const int ghalf = (kg >> 1) & 1;                          // fragment half
    const int e0    = ((kg & 1) * 4) + ((kg >= 4) ? 8 : 0);   // element base
    const int spos  = (grow + 16 * ghalf) * 16 + e0;          // LDS half-index

    const int gn = ntile * 16 + grow;
    const int gb = gn >> 10, gh = (gn >> 5) & 31, gw = gn & 31;
    const int gxbase = ((gb * INPUT_DIM) * HH + gh) * WW + gw;   // + i*HH*WW

    // ---- consumer role ----
    const int jcol = wave * 16 + mrow;
    const _Float16* bwave = bp + (size_t)wave * 512 + (size_t)lane * 16;

    const float INV2PI = 0.15915494309189535f;
    v8f acc = {};

    for (int i = 0; i < INPUT_DIM; ++i) {
        const float xt = x[gxbase + i * (HH * WW)] * INV2PI;  // revolutions/g
        float ph0 = xt * (float)(K0 + 1);
        float ph1 = ph0 + xt;
        float ph2 = ph1 + xt;
        float ph3 = ph2 + xt;
        const float dph = 32.0f * xt;
        const _Float16* bi = bwave + (size_t)i * (NCHUNK * 4096);

        for (int c = 0; c < NCHUNK; ++c) {
            const int buf = c & 1;
            // generate this chunk's A fragments (tail g>=300: B is 0, A don't-care)
            v4h ac, as;
            ac[0] = (_Float16)__builtin_amdgcn_cosf(ph0);
            ac[1] = (_Float16)__builtin_amdgcn_cosf(ph1);
            ac[2] = (_Float16)__builtin_amdgcn_cosf(ph2);
            ac[3] = (_Float16)__builtin_amdgcn_cosf(ph3);
            as[0] = (_Float16)__builtin_amdgcn_sinf(ph0);
            as[1] = (_Float16)__builtin_amdgcn_sinf(ph1);
            as[2] = (_Float16)__builtin_amdgcn_sinf(ph2);
            as[3] = (_Float16)__builtin_amdgcn_sinf(ph3);
            *(v4h*)&lds_a[buf][0][spos] = ac;
            *(v4h*)&lds_a[buf][1][spos] = as;
            ph0 += dph; ph1 += dph; ph2 += dph; ph3 += dph;
            __syncthreads();

            // consume: A from LDS, B from packed f16, 2 WMMAs (cos, sin)
            const v16h a0 = *(const v16h*)&lds_a[buf][0][lane * 16];
            const v16h a1 = *(const v16h*)&lds_a[buf][1][lane * 16];
            const v16h b0 = *(const v16h*)(bi);            // t=0 (cos coeffs)
            const v16h b1 = *(const v16h*)(bi + 2048);     // t=1 (sin coeffs)
            acc = __builtin_amdgcn_wmma_f32_16x16x32_f16(
                    false, a0, false, b0, (short)0, acc, false, false);
            acc = __builtin_amdgcn_wmma_f32_16x16x32_f16(
                    false, a1, false, b1, (short)0, acc, false, false);
            bi += 4096;
        }
    }

    // D layout: VGPR r -> row M = r + 8*half, col N = lane&15
    const float fw = fwp[0];
    #pragma unroll
    for (int r = 0; r < 8; ++r) {
        const int m = r + (half ? 8 : 0);
        const int n = ntile * 16 + m;
        const int b = n >> 10;
        const int h = (n >> 5) & 31;
        const int w = n & 31;
        combined[(((size_t)b * TOTAL_CH + jcol) * HH + h) * WW + w] = fw * acc[r];
    }
}

// ---------------------------------------------------------------------------
// Kernel 2: RBF branch -> channels [64, 320) of combined (NCHW).
// ---------------------------------------------------------------------------
__global__ __launch_bounds__(256) void rbf_kernel(
    const float* __restrict__ x, const float* __restrict__ rwp,
    float* __restrict__ combined)
{
    const int tid = blockIdx.x * blockDim.x + threadIdx.x; // indexes x directly
    if (tid >= BB * INPUT_DIM * HH * WW) return;
    const float xv = x[tid];
    const int w = tid & 31;
    const int h = (tid >> 5) & 31;
    const int i = (tid >> 10) & 31;
    const int b = tid >> 15;
    const float rw    = rwp[0];
    const float denom = 4.0f / (float)(RBF_GRIDS - 1);
    float* outb = combined +
        (((size_t)b * TOTAL_CH + OUTPUT_DIM + i * RBF_GRIDS) * HH + h) * WW + w;
    #pragma unroll
    for (int k = 0; k < RBF_GRIDS; ++k) {
        const float gk = -2.0f + denom * (float)k;
        const float d  = (xv - gk) / denom;
        outb[(size_t)k * HH * WW] = rw * __expf(-d * d);
    }
}

// ---------------------------------------------------------------------------
// Kernel 3: BatchNorm batch statistics -> per-channel affine (scale, shift).
// ---------------------------------------------------------------------------
__global__ __launch_bounds__(256) void bn_stats_kernel(
    const float* __restrict__ combined,
    const float* __restrict__ bnw, const float* __restrict__ bnb,
    float* __restrict__ scale, float* __restrict__ shift)
{
    const int c = blockIdx.x;
    __shared__ float s_sum[256];
    __shared__ float s_sq[256];
    float s = 0.0f, s2 = 0.0f;
    for (int e = threadIdx.x; e < NPTS; e += 256) {
        const int b  = e >> 10;
        const int hw = e & 1023;
        const float v = combined[((size_t)b * TOTAL_CH + c) * (HH * WW) + hw];
        s  += v;
        s2 += v * v;
    }
    s_sum[threadIdx.x] = s;
    s_sq[threadIdx.x]  = s2;
    __syncthreads();
    for (int off = 128; off > 0; off >>= 1) {
        if ((int)threadIdx.x < off) {
            s_sum[threadIdx.x] += s_sum[threadIdx.x + off];
            s_sq[threadIdx.x]  += s_sq[threadIdx.x + off];
        }
        __syncthreads();
    }
    if (threadIdx.x == 0) {
        const float mean = s_sum[0] / (float)NPTS;
        const float var  = s_sq[0] / (float)NPTS - mean * mean;
        const float sc   = bnw[c] * rsqrtf(var + BN_EPS);
        scale[c] = sc;
        shift[c] = bnb[c] - mean * sc;
    }
}

// ---------------------------------------------------------------------------
// Kernel 4: fused BN-apply + grouped 3x3 conv (pad=1, groups=4) + ReLU.
// ---------------------------------------------------------------------------
__global__ __launch_bounds__(256) void conv_kernel(
    const float* __restrict__ combined, const float* __restrict__ cw,
    const float* __restrict__ scale, const float* __restrict__ shift,
    float* __restrict__ out)
{
    const int hc = blockIdx.x & 3;
    const int oc = (blockIdx.x >> 2) & 63;
    const int b  = blockIdx.x >> 8;
    const int w  = threadIdx.x & 31;
    const int h  = hc * 8 + (threadIdx.x >> 5);
    const int gi    = oc >> 4;
    const int cbase = gi * 80;

    __shared__ float s_w[720];
    __shared__ float s_sc[80];
    __shared__ float s_sh[80];
    for (int idx = threadIdx.x; idx < 720; idx += 256)
        s_w[idx] = cw[(size_t)oc * 720 + idx];
    if (threadIdx.x < 80) {
        s_sc[threadIdx.x] = scale[cbase + threadIdx.x];
        s_sh[threadIdx.x] = shift[cbase + threadIdx.x];
    }
    __syncthreads();

    float acc = 0.0f;
    for (int ic = 0; ic < 80; ++ic) {
        const float sc = s_sc[ic];
        const float sh = s_sh[ic];
        const float* inp = combined + ((size_t)b * TOTAL_CH + cbase + ic) * (HH * WW);
        const float* wv  = s_w + ic * 9;
        #pragma unroll
        for (int kh = 0; kh < 3; ++kh) {
            const int hh = h + kh - 1;
            if (hh < 0 || hh >= HH) continue;
            #pragma unroll
            for (int kw = 0; kw < 3; ++kw) {
                const int ww_ = w + kw - 1;
                if (ww_ < 0 || ww_ >= WW) continue;
                acc += wv[kh * 3 + kw] * (inp[hh * WW + ww_] * sc + sh);
            }
        }
    }
    out[(((size_t)b * OUTPUT_DIM + oc) * HH + h) * WW + w] = fmaxf(acc, 0.0f);
}

// ---------------------------------------------------------------------------
extern "C" void kernel_launch(void* const* d_in, const int* in_sizes, int n_in,
                              void* d_out, int out_size, void* d_ws, size_t ws_size,
                              hipStream_t stream) {
    const float* x   = (const float*)d_in[0];  // (4,32,32,32)
    const float* fc  = (const float*)d_in[1];  // (2,64,32,300)
    const float* cw  = (const float*)d_in[2];  // (64,80,3,3)
    const float* bnw = (const float*)d_in[3];  // (320,)
    const float* bnb = (const float*)d_in[4];  // (320,)
    const float* fwp = (const float*)d_in[5];  // (1,)
    const float* rwp = (const float*)d_in[6];  // (1,)
    float* out = (float*)d_out;                // (4,64,32,32)

    const size_t CMB = (size_t)BB * TOTAL_CH * HH * WW;       // 1,310,720 f32
    float* combined = (float*)d_ws;
    float* scale    = combined + CMB;                         // 320 f32
    float* shift    = scale + TOTAL_CH;                       // 320 f32
    _Float16* packed = (_Float16*)(shift + TOTAL_CH);         // 2.62 MB, 16B-aligned

    pack_b_kernel<<<(int)((PACK_ELEMS + 255) / 256), 256, 0, stream>>>(fc, packed);
    fourier_wmma_kernel<<<256, 128, 0, stream>>>(x, packed, fwp, combined);
    rbf_kernel<<<(BB * INPUT_DIM * HH * WW) / 256, 256, 0, stream>>>(x, rwp, combined);
    bn_stats_kernel<<<TOTAL_CH, 256, 0, stream>>>(combined, bnw, bnb, scale, shift);
    conv_kernel<<<BB * OUTPUT_DIM * 4, 256, 0, stream>>>(combined, cw, scale, shift, out);
}
